// Sae_90151363543076
// MI455X (gfx1250) — matrix operand order repaired and checked by reading, not compiled, not run on hardware
//
#include <hip/hip_runtime.h>
#include <stdint.h>

#define D_IN   2048
#define D_SAE  32768
#define N_TOK  8192
#define K_TOP  64

typedef __bf16 v16bf __attribute__((ext_vector_type(16)));
typedef float  v8f   __attribute__((ext_vector_type(8)));
typedef unsigned int u32x4 __attribute__((ext_vector_type(4)));
typedef int i32x4 __attribute__((ext_vector_type(4)));
typedef int i32x8 __attribute__((ext_vector_type(8)));

union FragU { uint4 q[2]; v16bf v; };

#if defined(__has_builtin)
#  if __has_builtin(__builtin_amdgcn_tensor_load_to_lds)
#    define HAVE_TDM 1
#  else
#    define HAVE_TDM 0
#  endif
#  if __has_builtin(__builtin_amdgcn_s_wait_tensorcnt)
#    define WAIT_TENSOR(n) __builtin_amdgcn_s_wait_tensorcnt(n)
#  else
#    define WAIT_TENSOR(n) asm volatile("s_wait_tensorcnt %0" :: "i"(n))
#  endif
#else
#  define HAVE_TDM 0
#  define WAIT_TENSOR(n) asm volatile("s_wait_tensorcnt %0" :: "i"(n))
#endif

// Compiler-level memory clobber: LDS is written by the TDM behind the
// compiler's back, so stores must not be forwarded / loads not CSE'd across
// stage publishes.
#define LDS_MUTATED() asm volatile("" ::: "memory")

// round-to-nearest-even fp32 -> bf16, packed pair (a low 16, b high 16)
__device__ __forceinline__ uint32_t pk_bf16(float a, float b) {
    uint32_t ua = __float_as_uint(a);
    uint32_t ub = __float_as_uint(b);
    ua += 0x7FFFu + ((ua >> 16) & 1u);
    ub += 0x7FFFu + ((ub >> 16) & 1u);
    return (ua >> 16) | (ub & 0xFFFF0000u);
}

// ---------------------------------------------------------------------------
// Kernel 0: one-shot fp32 -> bf16 conversion (off the GEMM hot path).
// One uint4 (8 bf16) out per thread.
// ---------------------------------------------------------------------------
__global__ __launch_bounds__(256)
void cvt_f32_to_bf16(const float* __restrict__ src, uint32_t* __restrict__ dst) {
    size_t i = (size_t)blockIdx.x * 256 + threadIdx.x;
    const float4* s = (const float4*)src + i * 2;
    float4 f0 = s[0], f1 = s[1];
    uint4 q;
    q.x = pk_bf16(f0.x, f0.y); q.y = pk_bf16(f0.z, f0.w);
    q.z = pk_bf16(f1.x, f1.y); q.w = pk_bf16(f1.z, f1.w);
    ((uint4*)dst)[i] = q;
}

// ---------------------------------------------------------------------------
// Kernel 1: acts = relu(x @ W_enc + b_enc)  via v_wmma_f32_16x16x32_bf16
// Block tile 128(M) x 128(N), 256 threads = 8 wave32 in 2(M) x 4(N).
// Each wave: 64x32 tile = 4 M-frags x 2 N-frags, fp32 accum.
// LDS: A rows padded to 80B (20 dwords), B rows padded to 272B (68 dwords)
//      -> conflict-free b128 fragment loads, double buffered.
// Tiles staged by the Tensor Data Mover (tensor_load_to_lds) with D# padding
// producing the padded layout directly; wave 0 issues, s_wait_tensorcnt +
// workgroup barrier publishes the stage.
// ---------------------------------------------------------------------------
#define A_STRIDE_U 20u   // dwords per A row (32 bf16 = 16 dw + 4 dw pad)
#define B_STRIDE_U 68u   // dwords per B row (128 bf16 = 64 dw + 4 dw pad)
#define A_BUF_U (128u * A_STRIDE_U)   // 2560 dw
#define B_BUF_U (32u  * B_STRIDE_U)   // 2176 dw
#define STAGE_U (A_BUF_U + B_BUF_U)   // 4736 dw

#if HAVE_TDM
// Issue one 2D-tile TDM load (bf16 elements) into LDS (this kernel's only
// __shared__ array sits at LDS offset 0, so descriptor lds_addr == byte
// offset into `lds`). pad fields: interval code (1<<c)*8 bytes, amount code
// (c+1) dwords appended per interval.
__device__ __forceinline__ void tdm_load_tile(const void* gptr, uint32_t ldsOff,
                                              uint32_t tdim0, uint32_t tdim1,
                                              uint32_t stride0,
                                              uint32_t tile0, uint32_t tile1,
                                              uint32_t padIv, uint32_t padAmt) {
    uint64_t ga = (uint64_t)(uintptr_t)gptr;
    u32x4 g0;
    g0[0] = 1u;                                                // count=1, no gather
    g0[1] = ldsOff;                                            // lds_addr (bytes)
    g0[2] = (uint32_t)ga;                                      // global_addr lo
    g0[3] = ((uint32_t)(ga >> 32) & 0x01FFFFFFu) | (2u << 30); // hi + type=2
    i32x8 g1;
    g1[0] = (int)(0x00010000u |            // data_size=1 (2 bytes), wg_mask=0
                  (1u << 20) |             // pad_enable
                  (padIv << 22) | (padAmt << 25));
    g1[1] = (int)((tdim0 & 0xFFFFu) << 16);                    // tensor_dim0 lo16
    g1[2] = (int)((tdim0 >> 16) | ((tdim1 & 0xFFFFu) << 16));
    g1[3] = (int)((tdim1 >> 16) | (tile0 << 16));              // tile_dim0
    g1[4] = (int)tile1;                                        // tile_dim1 (tile_dim2=0)
    g1[5] = (int)stride0;                                      // tensor_dim0_stride lo32
    g1[6] = 0;
    g1[7] = 0;
    i32x4 z4 = {0, 0, 0, 0};
#if __clang_major__ >= 23
    i32x8 z8 = {0, 0, 0, 0, 0, 0, 0, 0};
    __builtin_amdgcn_tensor_load_to_lds(g0, g1, z4, z4, z8, 0);
#else
    __builtin_amdgcn_tensor_load_to_lds(g0, g1, z4, z4, 0);
#endif
}
#endif

__global__ __launch_bounds__(256)
void sae_enc_gemm(const uint16_t* __restrict__ x16,
                  const uint16_t* __restrict__ w16,
                  const float* __restrict__ b_enc,
                  float* __restrict__ acts) {
    __shared__ uint32_t lds[2u * STAGE_U];   // 37,888 B, at LDS offset 0

    const int t     = threadIdx.x;
    const int lane  = t & 31;
    const int wave  = t >> 5;
    const int waveM = wave >> 2;   // 0..1
    const int waveN = wave & 3;    // 0..3
    const int m0    = blockIdx.y * 128;
    const int n0    = blockIdx.x * 128;

    v8f acc[4][2];
    #pragma unroll
    for (int mi = 0; mi < 4; ++mi)
        #pragma unroll
        for (int ni = 0; ni < 2; ++ni)
            #pragma unroll
            for (int r = 0; r < 8; ++r) acc[mi][ni][r] = 0.0f;

    const int mlane = lane & 15;
    const int koffA = (lane < 16) ? 0 : 4;   // dwords: lanes>=16 start at K=8

    auto compute = [&](int buf) {
        const uint32_t* A = lds + (unsigned)buf * STAGE_U;
        const uint32_t* B = A + A_BUF_U;
        // B fragments: lane holds column n=lane&15, 16 k in table order
        FragU bf[2];
        const uint16_t* bp16 = (const uint16_t*)B + waveN * 32 + (lane & 15);
        const int hb = lane >> 4;
        #pragma unroll
        for (int ni = 0; ni < 2; ++ni) {
            #pragma unroll
            for (int s = 0; s < 16; ++s) {
                const int kk = (s < 8) ? (8 * hb + s) : (16 + 8 * hb + (s - 8));
                ((uint16_t*)bf[ni].q)[s] = bp16[kk * (2 * B_STRIDE_U) + ni * 16];
            }
        }
        #pragma unroll
        for (int mi = 0; mi < 4; ++mi) {
            const uint32_t* ap =
                A + (waveM * 64 + mi * 16 + mlane) * A_STRIDE_U + koffA;
            FragU af;
            af.q[0] = *(const uint4*)(ap);       // K 0..7   (or 8..15)
            af.q[1] = *(const uint4*)(ap + 8);   // K 16..23 (or 24..31)
            acc[mi][0] = __builtin_amdgcn_wmma_f32_16x16x32_bf16(
                false, af.v, false, bf[0].v, (short)0, acc[mi][0], false, false);
            acc[mi][1] = __builtin_amdgcn_wmma_f32_16x16x32_bf16(
                false, af.v, false, bf[1].v, (short)0, acc[mi][1], false, false);
        }
    };

    const int KT = D_IN / 32;   // 64

#if HAVE_TDM
    // One-time zero-init: the TDM writes LDS invisibly to the compiler; with
    // no visible store the LDS global would be "never initialized" and every
    // load from it folds to undef. These stores make the memory defined; the
    // in-loop clobber stops the zeros being store-forwarded into the frags.
    for (unsigned i = (unsigned)t; i < 2u * STAGE_U; i += 256u) lds[i] = 0u;
    __syncthreads();

    const uint16_t* Ag = x16 + (size_t)m0 * D_IN;   // A tile origin (row m0, k 0)
    const uint16_t* Bg = w16 + n0;                  // B tile origin (k 0, col n0)
    auto issue = [&](int buf, int kt) {
        const uint32_t stage = (uint32_t)buf * (STAGE_U * 4u);   // bytes
        // A: 128 rows x 32 bf16; row 64B, pad 16B -> 80B stride
        tdm_load_tile(Ag + kt * 32, stage,
                      D_IN, 128u, D_IN, 32u, 128u, /*padIv=*/3u, /*padAmt=*/3u);
        // B: 32 rows x 128 bf16; row 256B, pad 16B -> 272B stride
        tdm_load_tile(Bg + (size_t)kt * 32 * D_SAE, stage + A_BUF_U * 4u,
                      D_SAE, 32u, D_SAE, 128u, 32u, /*padIv=*/5u, /*padAmt=*/3u);
    };
    if (wave == 0) issue(0, 0);
    for (int kt = 0; kt < KT; ++kt) {
        const int cur = kt & 1;
        if (wave == 0) {
            if (kt + 1 < KT) { issue(cur ^ 1, kt + 1); WAIT_TENSOR(2); }
            else             { WAIT_TENSOR(0); }
        }
        __syncthreads();          // publish stage `cur` to all waves
        LDS_MUTATED();            // TDM wrote LDS: loads must re-read memory
        compute(cur);
        __syncthreads();          // all readers done before stage is overwritten
    }
#else
    // Fallback: cooperative bf16 global loads + ds_store_b128, double buffered.
    const int arow = t >> 1, ahalf = t & 1;        // A: 128 rows x 2 x 32B
    const int brow = t >> 3, bseg  = t & 7;        // B: 32 rows x 8 x 32B
    const uint16_t* aptr = x16 + (size_t)(m0 + arow) * D_IN + ahalf * 16;
    const uint16_t* bptr = w16 + (size_t)brow * D_SAE + n0 + bseg * 16;
    uint4 ra[2], rb[2];
    auto gload = [&](int kt) {
        const uint4* pa = (const uint4*)(aptr + kt * 32);
        const uint4* pb = (const uint4*)(bptr + (size_t)kt * 32 * D_SAE);
        ra[0] = pa[0]; ra[1] = pa[1];
        rb[0] = pb[0]; rb[1] = pb[1];
    };
    auto lstore = [&](int buf) {
        uint32_t* da = lds + (unsigned)buf * STAGE_U + arow * A_STRIDE_U + ahalf * 8;
        *(uint4*)(da) = ra[0]; *(uint4*)(da + 4) = ra[1];
        uint32_t* db = lds + (unsigned)buf * STAGE_U + A_BUF_U + brow * B_STRIDE_U + bseg * 8;
        *(uint4*)(db) = rb[0]; *(uint4*)(db + 4) = rb[1];
    };
    gload(0); lstore(0);
    __syncthreads();
    for (int kt = 0; kt < KT; ++kt) {
        const int cur = kt & 1;
        if (kt + 1 < KT) gload(kt + 1);
        compute(cur);
        if (kt + 1 < KT) { lstore(cur ^ 1); __syncthreads(); }
    }
#endif

    // Epilogue: D layout — VGPR r -> M = r (+8 for lanes 16..31), N = lane&15
    const int nIdx = lane & 15;
    const int mAdd = (lane >> 4) << 3;
    #pragma unroll
    for (int ni = 0; ni < 2; ++ni) {
        const int gcol = n0 + waveN * 32 + ni * 16 + nIdx;
        const float bias = b_enc[gcol];
        #pragma unroll
        for (int mi = 0; mi < 4; ++mi) {
            const int growB = m0 + waveM * 64 + mi * 16 + mAdd;
            v8f c = acc[mi][ni];
            #pragma unroll
            for (int r = 0; r < 8; ++r) {
                float v = c[r] + bias;
                v = v > 0.0f ? v : 0.0f;
                acts[(size_t)(growB + r) * D_SAE + gcol] = v;
            }
        }
    }
}

// ---------------------------------------------------------------------------
// Kernel 2: per-row top-64 via radix threshold (values are >= 0 post-ReLU,
// so fp32 bit patterns are order-isomorphic to the values).
// ---------------------------------------------------------------------------
__global__ __launch_bounds__(256)
void sae_topk(const float* __restrict__ acts,
              float* __restrict__ top_vals,
              int* __restrict__ top_idx) {
    const int row = blockIdx.x;
    const float* a = acts + (size_t)row * D_SAE;
    const int t = threadIdx.x;

    __shared__ uint32_t hist[1024];
    __shared__ float svals[K_TOP];
    __shared__ int   sidx[K_TOP];
    __shared__ int   s_T, s_cnt;

    for (int i = t; i < 1024; i += 256) hist[i] = 0u;
    if (t == 0) s_cnt = 0;
    __syncthreads();

    for (int i = t; i < D_SAE; i += 256) {
        uint32_t b = __float_as_uint(a[i]) >> 21;
        atomicAdd(&hist[b], 1u);
    }
    __syncthreads();

    if (t == 0) {
        uint32_t cum = 0; int T = 0;
        for (int b = 1023; b >= 0; --b) {
            uint32_t c = hist[b];
            if (cum + c >= (uint32_t)K_TOP) { T = b; break; }
            cum += c;
        }
        s_T = T;
    }
    __syncthreads();
    const int T = s_T;

    for (int i = t; i < D_SAE; i += 256) {
        float v = a[i];
        int b = (int)(__float_as_uint(v) >> 21);
        if (b > T) {
            int p = atomicAdd(&s_cnt, 1);
            svals[p] = v; sidx[p] = i;
        }
    }
    __syncthreads();
    for (int i = t; i < D_SAE; i += 256) {
        float v = a[i];
        int b = (int)(__float_as_uint(v) >> 21);
        if (b == T) {
            int p = atomicAdd(&s_cnt, 1);
            if (p < K_TOP) { svals[p] = v; sidx[p] = i; }
        }
    }
    __syncthreads();

    for (int ph = 0; ph < K_TOP; ++ph) {
        if (t < 32) {
            int j = (t << 1) | (ph & 1);
            if (j + 1 < K_TOP) {
                float v0 = svals[j], v1 = svals[j + 1];
                if (v0 < v1) {
                    svals[j] = v1; svals[j + 1] = v0;
                    int i0 = sidx[j]; sidx[j] = sidx[j + 1]; sidx[j + 1] = i0;
                }
            }
        }
        __syncthreads();
    }

    if (t < K_TOP) {
        top_vals[row * K_TOP + t] = svals[t];
        top_idx [row * K_TOP + t] = sidx[t];
    }
}

// ---------------------------------------------------------------------------
// Kernel 3: y[row,:] = b_dec + sum_j top_vals[row,j] * W_dec[top_idx[row,j],:]
// ---------------------------------------------------------------------------
__global__ __launch_bounds__(256)
void sae_decode(const float* __restrict__ top_vals,
                const int* __restrict__ top_idx,
                const float* __restrict__ W_dec,
                const float* __restrict__ b_dec,
                float* __restrict__ y) {
    const int row = blockIdx.x;
    const int t = threadIdx.x;
    __shared__ float sv[K_TOP];
    __shared__ int   si[K_TOP];
    if (t < K_TOP) {
        sv[t] = top_vals[row * K_TOP + t];
        si[t] = top_idx [row * K_TOP + t];
    }
    __syncthreads();

    const int c0 = t * 8;   // 256 threads * 8 cols = 2048
    float4 acc0 = *(const float4*)(b_dec + c0);
    float4 acc1 = *(const float4*)(b_dec + c0 + 4);

    #pragma unroll 4
    for (int j = 0; j < K_TOP; ++j) {
        float v = sv[j];
        if (j + 1 < K_TOP)
            __builtin_prefetch(W_dec + (size_t)si[j + 1] * D_IN + c0, 0, 3);
        if (v != 0.0f) {
            const float* w = W_dec + (size_t)si[j] * D_IN + c0;
            float4 w0 = *(const float4*)(w);
            float4 w1 = *(const float4*)(w + 4);
            acc0.x += v * w0.x; acc0.y += v * w0.y;
            acc0.z += v * w0.z; acc0.w += v * w0.w;
            acc1.x += v * w1.x; acc1.y += v * w1.y;
            acc1.z += v * w1.z; acc1.w += v * w1.w;
        }
    }
    *(float4*)(y + (size_t)row * D_IN + c0)     = acc0;
    *(float4*)(y + (size_t)row * D_IN + c0 + 4) = acc1;
}

// ---------------------------------------------------------------------------
extern "C" void kernel_launch(void* const* d_in, const int* in_sizes, int n_in,
                              void* d_out, int out_size, void* d_ws, size_t ws_size,
                              hipStream_t stream) {
    const float* x     = (const float*)d_in[0];
    const float* W_enc = (const float*)d_in[1];
    const float* b_enc = (const float*)d_in[2];
    const float* W_dec = (const float*)d_in[3];
    const float* b_dec = (const float*)d_in[4];
    // d_in[5] = k (== 64), compile-time constant here

    float* y        = (float*)d_out;
    float* top_vals = y + (size_t)N_TOK * D_IN;
    int*   top_idx  = (int*)(top_vals + (size_t)N_TOK * K_TOP);

    // workspace layout: x16 (32 MB) | w16 (128 MB) | acts (1 GiB)
    uint16_t* x16  = (uint16_t*)d_ws;
    uint16_t* w16  = x16 + (size_t)N_TOK * D_IN;
    float*    acts = (float*)(w16 + (size_t)D_IN * D_SAE);

    // one uint4 (8 elements) per thread
    const int xblk = (int)(((size_t)N_TOK * D_IN / 8) / 256);   // 8192
    const int wblk = (int)(((size_t)D_IN * D_SAE / 8) / 256);   // 32768
    cvt_f32_to_bf16<<<xblk, 256, 0, stream>>>(x,     (uint32_t*)x16);
    cvt_f32_to_bf16<<<wblk, 256, 0, stream>>>(W_enc, (uint32_t*)w16);

    dim3 gemm_grid(D_SAE / 128, N_TOK / 128);   // 256 x 64 blocks
    sae_enc_gemm<<<gemm_grid, 256, 0, stream>>>(x16, w16, b_enc, acts);
    sae_topk  <<<N_TOK, 256, 0, stream>>>(acts, top_vals, top_idx);
    sae_decode<<<N_TOK, 256, 0, stream>>>(top_vals, top_idx, W_dec, b_dec, y);
}
